// BiLSTMCRF_31318901522848
// MI455X (gfx1250) — matrix operand order repaired and checked by reading, not compiled, not run on hardware
//
#include <hip/hip_runtime.h>
#include <hip/hip_bf16.h>

// ---------------------------------------------------------------------------
// BiLSTM-CRF for MI455X (gfx1250, wave32, WMMA f16 16x16x32 -> f32 acc)
// Recurrent scan: Whh staged in LDS (320KB WGP budget), gates fused in regs,
// double-buffered h16 -> one barrier per timestep on the serial path.
// ---------------------------------------------------------------------------

typedef _Float16 half_t;
typedef _Float16 v8h  __attribute__((ext_vector_type(8)));
typedef _Float16 v16h __attribute__((ext_vector_type(16)));
typedef float    v8f  __attribute__((ext_vector_type(8)));

#define Bn 64
#define Sn 512
#define Hn 128
#define G4 512      // 4*H
#define NTAG 25
#define CLn 16
#define CEn 30
#define CHn 50
#define WEn 100
#define Mrows (Sn*Bn)   // 32768

#define HBUF_BYTES (Bn * Hn * 2)            // 16 KB per h16 buffer
#define WHH_BYTES  (G4 * Hn * 2)            // 128 KB
#define LSTM_LDS_BYTES (2 * HBUF_BYTES + WHH_BYTES)   // 160 KB dynamic LDS

__device__ __forceinline__ float sigm(float x) { return 1.0f / (1.0f + __expf(-x)); }

__device__ __forceinline__ v8f wmma_f16(v16h a, v16h b, v8f c) {
    return __builtin_amdgcn_wmma_f32_16x16x32_f16(false, a, false, b, (short)0, c,
                                                  false, false);
}

// A fragment (16x32 f16, M x K), row-major source with leading dim ldk.
// Lane<16: row=mBase+lane,   K chunks {kb..kb+7},{kb+16..kb+23}
// Lane>=16: row=mBase+lane-16, K chunks {kb+8..kb+15},{kb+24..kb+31}
__device__ __forceinline__ v16h load_a_global(const half_t* __restrict__ base,
                                              int ldk, int mBase, int kb, int lane) {
    int row = mBase + (lane & 15);
    const half_t* p = base + (size_t)row * ldk + kb + ((lane < 16) ? 0 : 8);
    v8h c0 = *(const v8h*)p;
    v8h c1 = *(const v8h*)(p + 16);
    v16h a;
#pragma unroll
    for (int i = 0; i < 8; ++i) { a[i] = c0[i]; a[i + 8] = c1[i]; }
    return a;
}

// B fragment (32x16 f16, K x N) built from W[n][k] row-major (ldk): B[k][n]=W[n][k].
// Lane<16: col n=nBase+lane, K chunks {kb..kb+7},{kb+16..kb+23}; lane>=16: {kb+8..kb+15},{kb+24..kb+31}.
__device__ __forceinline__ v16h load_b_w(const half_t* __restrict__ W,
                                         int ldk, int nBase, int kb, int lane) {
    int n = nBase + (lane & 15);
    int k = kb + ((lane < 16) ? 0 : 8);
    const half_t* p = W + (size_t)n * ldk + k;
    v8h c0 = *(const v8h*)p;
    v8h c1 = *(const v8h*)(p + 16);
    v16h b;
#pragma unroll
    for (int i = 0; i < 8; ++i) { b[i] = c0[i]; b[i + 8] = c1[i]; }
    return b;
}

// ---------------------------------------------------------------------------
// Weight prep: f32 -> f16 with zero-padding of K dimension
// ---------------------------------------------------------------------------
__global__ void cvt_pad_f16(const float* __restrict__ src, half_t* __restrict__ dst,
                            int rows, int sc, int dc) {
    int i = blockIdx.x * blockDim.x + threadIdx.x;
    if (i >= rows * dc) return;
    int r = i / dc, c = i % dc;
    dst[i] = (c < sc) ? (half_t)src[r * sc + c] : (half_t)0.0f;
}

__global__ void bias_sum(const float* __restrict__ a, const float* __restrict__ b,
                         float* __restrict__ o, int n) {
    int i = blockIdx.x * blockDim.x + threadIdx.x;
    if (i < n) o[i] = a[i] + b[i];
}

// ---------------------------------------------------------------------------
// Embedding + char CNN (+relu+maxpool) -> feat16[row=s*64+b][160] (150 used)
// ---------------------------------------------------------------------------
__global__ __launch_bounds__(64) void embed_conv(
    const int* __restrict__ words, const int* __restrict__ chars,
    const float* __restrict__ wtab, const float* __restrict__ ctab,
    const float* __restrict__ convw, const float* __restrict__ convb,
    half_t* __restrict__ feat) {
    __shared__ float ce[CLn][CEn];
    int p = blockIdx.x;               // p = b*S + s (reference flatten order)
    int b = p >> 9, s = p & 511;
    int row = s * Bn + b;
    int tid = threadIdx.x;
    for (int v = tid; v < CLn * CEn; v += 64) {
        int l = v / CEn, e = v % CEn;
        int ci = chars[(size_t)p * CLn + l];
        ce[l][e] = ctab[(size_t)ci * CEn + e];
    }
    __syncthreads();
    half_t* fr = feat + (size_t)row * 160;
    int w = words[p];
    for (int k = tid; k < WEn; k += 64) fr[k] = (half_t)wtab[(size_t)w * WEn + k];
    if (tid < CHn) {
        int ch = tid;
        float m = -3.4e38f;
        for (int l = 0; l < CLn; ++l) {
            float a = convb[ch];
#pragma unroll
            for (int dk = 0; dk < 3; ++dk) {
                int ll = l + dk - 1;
                if (ll >= 0 && ll < CLn)
                    for (int e = 0; e < CEn; ++e)
                        a += ce[ll][e] * convw[(ch * CEn + e) * 3 + dk];
            }
            m = fmaxf(m, a);
        }
        m = fmaxf(m, 0.0f);
        fr[WEn + ch] = (half_t)m;
    }
    if (tid < 10) fr[150 + tid] = (half_t)0.0f;   // zero K-padding
}

// ---------------------------------------------------------------------------
// Batched input projection: out[dir][row][n] = A[row][:K] @ W[dir][n][:K] + bias
// grid = (M/16, 2), block = 256 (8 waves, wave w owns N-tiles w*64 .. w*64+48)
// ---------------------------------------------------------------------------
__global__ __launch_bounds__(256) void gemm_proj(
    const half_t* __restrict__ A, int lda, int K,
    const half_t* __restrict__ W, int ldw,
    const float* __restrict__ bias, float* __restrict__ out) {
    int dir = blockIdx.y;
    int mBase = blockIdx.x * 16;
    int wv = threadIdx.x >> 5, lane = threadIdx.x & 31;
    const half_t* Wd = W + (size_t)dir * G4 * ldw;
    const float* bd = bias + dir * G4;
    float* od = out + (size_t)dir * (size_t)gridDim.x * 16 * G4;
    int n0 = wv * 64;
    v8f acc[4] = {};
    for (int kb = 0; kb < K; kb += 32) {
        v16h a = load_a_global(A, lda, mBase, kb, lane);
        __builtin_prefetch(Wd + (size_t)(n0 + (lane & 15)) * ldw + kb + 32, 0, 3);
#pragma unroll
        for (int j = 0; j < 4; ++j) {
            v16h bfrag = load_b_w(Wd, ldw, n0 + j * 16, kb, lane);
            acc[j] = wmma_f16(a, bfrag, acc[j]);
        }
    }
    int colLo = lane & 15, hi = lane >> 4;
#pragma unroll
    for (int j = 0; j < 4; ++j) {
        int n = n0 + j * 16 + colLo;
        float bv = bd[n];
#pragma unroll
        for (int r = 0; r < 8; ++r) {
            int row = mBase + r + 8 * hi;
            od[(size_t)row * G4 + n] = acc[j][r] + bv;
        }
    }
}

// ---------------------------------------------------------------------------
// Persistent recurrent LSTM scan, one block per direction (512 thr = 16 waves).
// Whh lives in LDS (preloaded once); h16 is double-buffered in LDS; c and all
// four gates live in registers (wave owns i/f/g/o tiles of the same hidden
// block), so each timestep is: ds-loads -> 8x WMMA -> gatesIn stream -> VALU
// -> ds-store h -> ONE barrier.
// ---------------------------------------------------------------------------
__global__ __launch_bounds__(512) void lstm_rec(
    const float* __restrict__ gatesIn,   // [2][S*64][512]
    const half_t* __restrict__ Whh,      // [2][512][128] f16
    half_t* __restrict__ hcat) {         // [S*64][256], this dir at +dir*128
    extern __shared__ __align__(16) char smem[];
    half_t* hbuf0 = (half_t*)smem;                       // 16 KB
    half_t* hbuf1 = (half_t*)(smem + HBUF_BYTES);        // 16 KB
    half_t* whhL  = (half_t*)(smem + 2 * HBUF_BYTES);    // 128 KB

    int dir = blockIdx.x;
    int tid = threadIdx.x;
    int wv = tid >> 5, lane = tid & 31;

    // preload Whh (this direction) into LDS, zero the t=0 h buffer
    const half_t* Wd = Whh + (size_t)dir * G4 * Hn;
    for (int i = tid; i < (G4 * Hn) / 8; i += 512)
        ((v8h*)whhL)[i] = ((const v8h*)Wd)[i];
    for (int i = tid; i < (Bn * Hn) / 8; i += 512)
        ((v8h*)hbuf0)[i] = (v8h){};
    __syncthreads();

    const float* gIn = gatesIn + (size_t)dir * Mrows * G4;
    int mBase = (wv & 3) * 16;          // batch tile
    int hb0 = (wv >> 2) * 2;            // two 16-wide hidden blocks per wave
    int colLo = lane & 15, hi = lane >> 4;

    v8f creg[2] = {};                   // cell state, in registers

    for (int t = 0; t < Sn; ++t) {
        int s = dir ? (Sn - 1 - t) : t;
        const half_t* hRead = (t & 1) ? hbuf1 : hbuf0;
        half_t* hWrite = (t & 1) ? hbuf0 : hbuf1;

        v8f acc[2][4] = {};
        for (int kb = 0; kb < Hn; kb += 32) {
            // A fragment from LDS hRead (ldk = 128)
            const half_t* pa = &hRead[(mBase + colLo) * Hn + kb + ((lane < 16) ? 0 : 8)];
            v8h a0 = *(const v8h*)pa;
            v8h a1 = *(const v8h*)(pa + 16);
            v16h a;
#pragma unroll
            for (int i = 0; i < 8; ++i) { a[i] = a0[i]; a[i + 8] = a1[i]; }
#pragma unroll
            for (int q = 0; q < 2; ++q) {
#pragma unroll
                for (int g = 0; g < 4; ++g) {
                    int nBase = g * Hn + (hb0 + q) * 16;
                    const half_t* pb = whhL + (nBase + colLo) * Hn + kb +
                                       ((lane < 16) ? 0 : 8);
                    v8h b0 = *(const v8h*)pb;
                    v8h b1 = *(const v8h*)(pb + 16);
                    v16h bf;
#pragma unroll
                    for (int i = 0; i < 8; ++i) { bf[i] = b0[i]; bf[i + 8] = b1[i]; }
                    acc[q][g] = wmma_f16(a, bf, acc[q][g]);
                }
            }
        }

        // fuse gatesIn + LSTM pointwise update entirely in registers
        const float* gRow = gIn + (size_t)s * Bn * G4;
#pragma unroll
        for (int q = 0; q < 2; ++q) {
            int jcol = (hb0 + q) * 16 + colLo;   // hidden index j in [0,128)
#pragma unroll
            for (int r = 0; r < 8; ++r) {
                int brow = mBase + r + 8 * hi;   // batch index
                const float* gr = gRow + (size_t)brow * G4;
                float gi = acc[q][0][r] + gr[jcol];
                float gf = acc[q][1][r] + gr[Hn + jcol];
                float gg = acc[q][2][r] + gr[2 * Hn + jcol];
                float go = acc[q][3][r] + gr[3 * Hn + jcol];
                float cc = creg[q][r];
                cc = sigm(gf) * cc + sigm(gi) * tanhf(gg);
                float hh = sigm(go) * tanhf(cc);
                creg[q][r] = cc;
                half_t hh16 = (half_t)hh;
                hWrite[brow * Hn + jcol] = hh16;
                hcat[(size_t)(s * Bn + brow) * 256 + dir * Hn + jcol] = hh16;
            }
        }
        __syncthreads();
    }
}

// ---------------------------------------------------------------------------
// FC head: emissions[b][s][t] = hcat1[s*64+b][:256] . fc_w[t] + fc_b[t]
// ---------------------------------------------------------------------------
__global__ __launch_bounds__(32) void fc_head(const half_t* __restrict__ hc,
                                              const float* __restrict__ fcw,
                                              const float* __restrict__ fcb,
                                              float* __restrict__ em) {
    __shared__ float v[256];
    int row = blockIdx.x, tid = threadIdx.x;
    const half_t* hr = hc + (size_t)row * 256;
    for (int i = tid; i < 256; i += 32) v[i] = (float)hr[i];
    __syncthreads();
    if (tid < NTAG) {
        float sacc = fcb[tid];
        const float* wr = fcw + tid * 256;
        for (int k = 0; k < 256; ++k) sacc += v[k] * wr[k];
        int b = row & 63, sidx = row >> 6;
        em[((size_t)b * Sn + sidx) * NTAG + tid] = sacc;
    }
}

// ---------------------------------------------------------------------------
// Gold path score (one thread per batch row)
// ---------------------------------------------------------------------------
__global__ __launch_bounds__(64) void gold_score(
    const int* __restrict__ tags, const int* __restrict__ mask,
    const float* __restrict__ em, const float* __restrict__ trans,
    const float* __restrict__ startt, const float* __restrict__ endt,
    float* __restrict__ gold) {
    int b = threadIdx.x;
    const int* tg = tags + b * Sn;
    const int* mk = mask + b * Sn;
    int t0 = tg[0];
    float g = startt[t0] + em[((size_t)b * Sn) * NTAG + t0];
    int prev = t0, last = 0;
    for (int s = 1; s < Sn; ++s) {
        int tt = tg[s];
        if (mk[s]) {
            g += trans[prev * NTAG + tt] + em[((size_t)b * Sn + s) * NTAG + tt];
            last = s;
        }
        prev = tt;
    }
    g += endt[tg[last]];
    gold[b] = g;
}

// ---------------------------------------------------------------------------
// CRF forward algorithm (persistent block) + final mean loss
// ---------------------------------------------------------------------------
__global__ __launch_bounds__(256) void crf_forward(
    const float* __restrict__ em, const int* __restrict__ mask,
    const float* __restrict__ trans, const float* __restrict__ startt,
    const float* __restrict__ endt, const float* __restrict__ gold,
    float* __restrict__ out) {
    __shared__ float tr[NTAG * NTAG];
    __shared__ float alpha[Bn * NTAG];
    __shared__ float anew[Bn * NTAG];
    __shared__ float logz[Bn];
    int tid = threadIdx.x;
    for (int i = tid; i < NTAG * NTAG; i += 256) tr[i] = trans[i];
    for (int i = tid; i < Bn * NTAG; i += 256) {
        int b = i / NTAG, j = i % NTAG;
        alpha[i] = startt[j] + em[((size_t)b * Sn) * NTAG + j];
    }
    __syncthreads();
    for (int s = 1; s < Sn; ++s) {
        for (int i = tid; i < Bn * NTAG; i += 256) {
            int b = i / NTAG, j = i % NTAG;
            const float* ab = alpha + b * NTAG;
            float m = -3.4e38f;
            for (int k = 0; k < NTAG; ++k) m = fmaxf(m, ab[k] + tr[k * NTAG + j]);
            float ssum = 0.0f;
            for (int k = 0; k < NTAG; ++k) ssum += __expf(ab[k] + tr[k * NTAG + j] - m);
            float val = m + __logf(ssum) + em[((size_t)b * Sn + s) * NTAG + j];
            anew[i] = mask[b * Sn + s] ? val : ab[j];
        }
        __syncthreads();
        for (int i = tid; i < Bn * NTAG; i += 256) alpha[i] = anew[i];
        __syncthreads();
    }
    if (tid < Bn) {
        int b = tid;
        float m = -3.4e38f;
        for (int j = 0; j < NTAG; ++j) m = fmaxf(m, alpha[b * NTAG + j] + endt[j]);
        float ssum = 0.0f;
        for (int j = 0; j < NTAG; ++j) ssum += __expf(alpha[b * NTAG + j] + endt[j] - m);
        logz[b] = m + __logf(ssum);
    }
    __syncthreads();
    if (tid == 0) {
        float acc = 0.0f;
        for (int b = 0; b < Bn; ++b) acc += logz[b] - gold[b];
        out[0] = acc / (float)Bn;
    }
}

// ---------------------------------------------------------------------------
extern "C" void kernel_launch(void* const* d_in, const int* in_sizes, int n_in,
                              void* d_out, int out_size, void* d_ws, size_t ws_size,
                              hipStream_t stream) {
    (void)in_sizes; (void)n_in; (void)out_size; (void)ws_size;
    const int*   words = (const int*)d_in[0];
    const int*   chars = (const int*)d_in[1];
    const int*   tags  = (const int*)d_in[2];
    const int*   mask  = (const int*)d_in[3];
    const float* wtab  = (const float*)d_in[4];
    const float* ctab  = (const float*)d_in[5];
    const float* convw = (const float*)d_in[6];
    const float* convb = (const float*)d_in[7];
    const float* wih0  = (const float*)d_in[8];
    const float* whh0  = (const float*)d_in[9];
    const float* bih0  = (const float*)d_in[10];
    const float* bhh0  = (const float*)d_in[11];
    const float* wih1  = (const float*)d_in[12];
    const float* whh1  = (const float*)d_in[13];
    const float* bih1  = (const float*)d_in[14];
    const float* bhh1  = (const float*)d_in[15];
    const float* fcw   = (const float*)d_in[16];
    const float* fcb   = (const float*)d_in[17];
    const float* trans = (const float*)d_in[18];
    const float* stt   = (const float*)d_in[19];
    const float* ent   = (const float*)d_in[20];

    char* ws = (char*)d_ws;
    auto alloc = [&](size_t bytes) -> char* {
        char* p = ws;
        ws += (bytes + 255) & ~(size_t)255;
        return p;
    };
    half_t* feat16 = (half_t*)alloc((size_t)Mrows * 160 * 2);
    half_t* wih0h  = (half_t*)alloc((size_t)2 * G4 * 160 * 2);
    half_t* wih1h  = (half_t*)alloc((size_t)2 * G4 * 256 * 2);
    half_t* whh0h  = (half_t*)alloc((size_t)2 * G4 * Hn * 2);
    half_t* whh1h  = (half_t*)alloc((size_t)2 * G4 * Hn * 2);
    float*  bias0  = (float*)alloc((size_t)2 * G4 * 4);
    float*  bias1  = (float*)alloc((size_t)2 * G4 * 4);
    float*  gates  = (float*)alloc((size_t)2 * Mrows * G4 * 4);  // reused L0/L1
    half_t* hcat0  = (half_t*)alloc((size_t)Mrows * 256 * 2);
    half_t* hcat1  = (half_t*)alloc((size_t)Mrows * 256 * 2);
    float*  emis   = (float*)alloc((size_t)Bn * Sn * NTAG * 4);
    float*  gold   = (float*)alloc((size_t)Bn * 4);

    // weight prep
    cvt_pad_f16<<<(2 * G4 * 160 + 255) / 256, 256, 0, stream>>>(wih0, wih0h, 2 * G4, 150, 160);
    cvt_pad_f16<<<(2 * G4 * 256 + 255) / 256, 256, 0, stream>>>(wih1, wih1h, 2 * G4, 256, 256);
    cvt_pad_f16<<<(2 * G4 * Hn + 255) / 256, 256, 0, stream>>>(whh0, whh0h, 2 * G4, Hn, Hn);
    cvt_pad_f16<<<(2 * G4 * Hn + 255) / 256, 256, 0, stream>>>(whh1, whh1h, 2 * G4, Hn, Hn);
    bias_sum<<<(2 * G4 + 255) / 256, 256, 0, stream>>>(bih0, bhh0, bias0, 2 * G4);
    bias_sum<<<(2 * G4 + 255) / 256, 256, 0, stream>>>(bih1, bhh1, bias1, 2 * G4);

    // features
    embed_conv<<<Mrows, 64, 0, stream>>>(words, chars, wtab, ctab, convw, convb, feat16);

    // layer 0: batched input projection + recurrent scan
    gemm_proj<<<dim3(Mrows / 16, 2), 256, 0, stream>>>(feat16, 160, 160, wih0h, 160, bias0, gates);
    lstm_rec<<<2, 512, LSTM_LDS_BYTES, stream>>>(gates, whh0h, hcat0);

    // layer 1
    gemm_proj<<<dim3(Mrows / 16, 2), 256, 0, stream>>>(hcat0, 256, 256, wih1h, 256, bias1, gates);
    lstm_rec<<<2, 512, LSTM_LDS_BYTES, stream>>>(gates, whh1h, hcat1);

    // emissions + CRF
    fc_head<<<Mrows, 32, 0, stream>>>(hcat1, fcw, fcb, emis);
    gold_score<<<1, 64, 0, stream>>>(tags, mask, emis, trans, stt, ent, gold);
    crf_forward<<<1, 256, 0, stream>>>(emis, mask, trans, stt, ent, gold, (float*)d_out);
}
